// GCNForMIS_35089882808545
// MI455X (gfx1250) — matrix operand order repaired and checked by reading, not compiled, not run on hardware
//
#include <hip/hip_runtime.h>
#include <hip/hip_bf16.h>
#include <math.h>

// ---------------------------------------------------------------------------
// GCN forward on MI455X (gfx1250, wave32).
//   * H=128 GEMMs  -> v_wmma_f32_16x16x32_bf16 (bf16 in, f32 accumulate),
//                     LDS-staged operands fetched as ds_load_b128
//   * aggregation  -> wave-per-edge float4 gather + global_atomic_add_f32
//   * self-loops   -> separate pass that also initializes the accumulator
// ---------------------------------------------------------------------------

typedef __attribute__((ext_vector_type(16))) __bf16 v16bf;
typedef __attribute__((ext_vector_type(8)))  float  v8f;

union ABf { v16bf v; unsigned int u[8]; uint4 q[2]; };

#define LSTRIDE 68   // uints per row: 16B-aligned rows, 4-bank rotation -> conflict-free b128

__device__ __forceinline__ unsigned int pack_bf16x2(float a, float b) {
  unsigned int ua = __float_as_uint(a);
  unsigned int ub = __float_as_uint(b);
  ua = (ua + 0x7FFFu + ((ua >> 16) & 1u)) >> 16;   // RNE f32 -> bf16
  ub = (ub + 0x7FFFu + ((ub >> 16) & 1u)) >> 16;
  return (ua & 0xFFFFu) | (ub << 16);
}

// ---------------- small elementwise / graph kernels ------------------------

__global__ void gcn_fill_f32(float* p, float v, int n) {
  int i = blockIdx.x * blockDim.x + threadIdx.x;
  if (i < n) p[i] = v;
}

__global__ void gcn_deg_count(float* deg, const long long* __restrict__ dst, int E) {
  int e = blockIdx.x * blockDim.x + threadIdx.x;
  if (e < E) atomicAdd(&deg[(int)dst[e]], 1.0f);
}

__global__ void gcn_rsqrt_ip(float* d, int n) {
  int i = blockIdx.x * blockDim.x + threadIdx.x;
  if (i < n) {
    float v = d[i];
    d[i] = (v > 0.0f) ? rsqrtf(v) : 0.0f;
  }
}

// z[i][c] = x[i] * W0[c]   (layer-0 GEMM is an outer product, N x 1 @ 1 x 128)
__global__ void gcn_layer0(const float* __restrict__ x, const float* __restrict__ W0,
                           float* __restrict__ z, int n) {
  int idx = blockIdx.x * blockDim.x + threadIdx.x;
  if (idx < n * 128) z[idx] = x[idx >> 7] * W0[idx & 127];
}

// agg[i][:] = dis[i]^2 * z[i][:]   (self-loop term; initializes agg)
__global__ void gcn_agg_self(const float* __restrict__ z, float* __restrict__ agg,
                             const float* __restrict__ dis, int n) {
  int idx = blockIdx.x * blockDim.x + threadIdx.x;   // n*32 threads, float4 each
  int i = idx >> 5, c4 = (idx & 31);
  if (i >= n) return;
  float w = dis[i] * dis[i];
  float4 v = ((const float4*)(z + (size_t)i * 128))[c4];
  float4 o; o.x = w * v.x; o.y = w * v.y; o.z = w * v.z; o.w = w * v.w;
  ((float4*)(agg + (size_t)i * 128))[c4] = o;
}

// one wave32 per edge: gather z[src] (float4/lane), scatter-add into agg[dst]
__global__ void gcn_agg_edges(const float* __restrict__ z, float* __restrict__ agg,
                              const long long* __restrict__ src,
                              const long long* __restrict__ dst,
                              const float* __restrict__ dis, int E) {
  int gid = blockIdx.x * blockDim.x + threadIdx.x;
  int e = gid >> 5, lane = gid & 31;
  if (e >= E) return;
  int s = (int)src[e], d = (int)dst[e];
  float w = dis[s] * dis[d];
  float4 v = ((const float4*)(z + (size_t)s * 128))[lane];
  float* ad = agg + (size_t)d * 128 + lane * 4;
  atomicAdd(ad + 0, w * v.x);
  atomicAdd(ad + 1, w * v.y);
  atomicAdd(ad + 2, w * v.z);
  atomicAdd(ad + 3, w * v.w);
}

__global__ void gcn_bias_relu(float* __restrict__ h, const float* __restrict__ b, int n) {
  int idx = blockIdx.x * blockDim.x + threadIdx.x;
  if (idx < n * 128) h[idx] = fmaxf(h[idx] + b[idx & 127], 0.0f);
}

// zo[i] = dot(h[i,:], Wo)  -- one wave per node, float4 per lane + shuffle tree
__global__ void gcn_dot_out(const float* __restrict__ h, const float* __restrict__ Wo,
                            float* __restrict__ zo, int n) {
  int gid = blockIdx.x * blockDim.x + threadIdx.x;
  int i = gid >> 5, lane = gid & 31;
  if (i >= n) return;
  float4 hv = ((const float4*)(h + (size_t)i * 128))[lane];
  float4 wv = ((const float4*)Wo)[lane];
  float p = hv.x * wv.x + hv.y * wv.y + hv.z * wv.z + hv.w * wv.w;
  for (int off = 16; off >= 1; off >>= 1) p += __shfl_xor(p, off);
  if (lane == 0) zo[i] = p;
}

__global__ void gcn_acc_self(const float* __restrict__ zo, float* __restrict__ acc,
                             const float* __restrict__ dis, int n) {
  int i = blockIdx.x * blockDim.x + threadIdx.x;
  if (i < n) acc[i] = dis[i] * dis[i] * zo[i];
}

__global__ void gcn_acc_edges(const float* __restrict__ zo, float* __restrict__ acc,
                              const long long* __restrict__ src,
                              const long long* __restrict__ dst,
                              const float* __restrict__ dis, int E) {
  int e = blockIdx.x * blockDim.x + threadIdx.x;
  if (e >= E) return;
  int s = (int)src[e], d = (int)dst[e];
  atomicAdd(&acc[d], dis[s] * dis[d] * zo[s]);
}

__global__ void gcn_sigmoid(const float* __restrict__ acc, const float* __restrict__ bo,
                            float* __restrict__ out, int n) {
  int i = blockIdx.x * blockDim.x + threadIdx.x;
  if (i < n) out[i] = 1.0f / (1.0f + expf(-(acc[i] + bo[0])));
}

// ---------------- WMMA GEMM: out[N,128] = A[N,128] @ W[128,128] -------------
// 256 threads = 8 wave32 per block; block owns 128 output rows x all 128 cols.
// A tile + W staged as packed bf16 pairs in LDS. Row stride 68 uints:
// rows are 16B-aligned and rotate 4 banks/row, so each per-lane ds_load_b128
// covers banks 4l..4l+3 -> 16 lanes tile all 64 banks conflict-free.
// Each wave: 8 accumulator tiles (v8f), K-loop = 4 x (K=32) WMMA.
__global__ void __launch_bounds__(256)
gcn_gemm_wmma(const float* __restrict__ A, const float* __restrict__ W,
              float* __restrict__ out, int n) {
  __shared__ __align__(16) unsigned int lA[128 * LSTRIDE];  // [row][kpair]
  __shared__ __align__(16) unsigned int lB[128 * LSTRIDE];  // [col][kpair] (W^T pairs)

  const int tid = threadIdx.x;
  const int mbase = blockIdx.x * 128;

  // Stage A tile: float4 global loads, pack to bf16 pairs, uint2 LDS stores.
  for (int idx = tid; idx < 128 * 32; idx += 256) {
    int r = idx >> 5, q = idx & 31;            // q: float4 index within row
    int row = mbase + r;
    float4 f = make_float4(0.f, 0.f, 0.f, 0.f);
    if (row < n) f = ((const float4*)(A + (size_t)row * 128))[q];
    uint2 p;
    p.x = pack_bf16x2(f.x, f.y);
    p.y = pack_bf16x2(f.z, f.w);
    *(uint2*)&lA[r * LSTRIDE + q * 2] = p;
  }
  // Stage W transposed: lB[col][kp] = pack(W[2kp][col], W[2kp+1][col]).
  for (int idx = tid; idx < 128 * 64; idx += 256) {
    int c = idx & 127, kp = idx >> 7;
    float f0 = W[(size_t)(kp * 2) * 128 + c];
    float f1 = W[(size_t)(kp * 2 + 1) * 128 + c];
    lB[c * LSTRIDE + kp] = pack_bf16x2(f0, f1);
  }
  __syncthreads();

  const int wave  = tid >> 5;
  const int lane  = tid & 31;
  const int l15   = lane & 15;
  const int khalf = lane >> 4;

  v8f acc8[8];
#pragma unroll
  for (int ct = 0; ct < 8; ++ct) acc8[ct] = (v8f){0,0,0,0,0,0,0,0};

  const int mrow = (wave << 4) + l15;   // A lane row (ISA: lanes 0-15 == 16-31 in M)

#pragma unroll
  for (int kb = 0; kb < 4; ++kb) {
    // A operand (ISA 16-bit A 16x32): VGPRs 0-3 -> kpair base+{0..3},
    // VGPRs 4-7 -> kpair base+8+{0..3}, base = kb*16 + khalf*4. Two b128 loads.
    ABf a;
    {
      const unsigned int* ap = &lA[mrow * LSTRIDE + (kb << 4) + (khalf << 2)];
      a.q[0] = *(const uint4*)(ap);
      a.q[1] = *(const uint4*)(ap + 8);
    }
#pragma unroll
    for (int ct = 0; ct < 8; ++ct) {
      // B operand (32x16): same K order as A: VGPRs 0-3 -> kpair base+{0..3},
      // VGPRs 4-7 -> kpair base+8+{0..3}, base = kb*16 + khalf*4.
      ABf b;
      const unsigned int* bp = &lB[((ct << 4) + l15) * LSTRIDE + (kb << 4) + (khalf << 2)];
      b.q[0] = *(const uint4*)(bp);
      b.q[1] = *(const uint4*)(bp + 8);
      acc8[ct] = __builtin_amdgcn_wmma_f32_16x16x32_bf16(
          false, a.v, false, b.v, (short)0, acc8[ct], false, false);
    }
  }

  // Store: C/D layout — lanes 0-15: N=lane, VGPR r: M=r; lanes 16-31: M=r+8.
  const int rbase = mbase + (wave << 4) + ((lane < 16) ? 0 : 8);
#pragma unroll
  for (int ct = 0; ct < 8; ++ct) {
#pragma unroll
    for (int r = 0; r < 8; ++r) {
      int row = rbase + r;
      if (row < n) out[(size_t)row * 128 + (ct << 4) + l15] = acc8[ct][r];
    }
  }
}

// ---------------------------------------------------------------------------

extern "C" void kernel_launch(void* const* d_in, const int* in_sizes, int n_in,
                              void* d_out, int out_size, void* d_ws, size_t ws_size,
                              hipStream_t stream) {
  const float*     x   = (const float*)d_in[0];
  const long long* ei  = (const long long*)d_in[1];
  const float*     W0  = (const float*)d_in[2];
  const float*     b0  = (const float*)d_in[3];
  const float*     W1  = (const float*)d_in[4];
  const float*     b1  = (const float*)d_in[5];
  const float*     W2  = (const float*)d_in[6];
  const float*     b2  = (const float*)d_in[7];
  const float*     Wo  = (const float*)d_in[8];
  const float*     bo  = (const float*)d_in[9];
  float*           out = (float*)d_out;

  const int n = in_sizes[0];          // N nodes (x is [N,1])
  const int E = in_sizes[1] / 2;      // edge_index is [2,E]
  const long long* src = ei;
  const long long* dst = ei + E;

  const size_t NH = (size_t)n * 128;
  float* wsA = (float*)d_ws;          // z buffers ping-pong
  float* wsB = wsA + NH;
  float* dis = wsB + NH;              // deg -> dis (in place)
  float* zo  = dis + n;
  float* acc = zo + n;

  const int T = 256;
  const int gN   = (n + T - 1) / T;
  const int gE   = (E + T - 1) / T;
  const int gNH  = (int)((NH + T - 1) / T);
  const int gN32 = (int)(((size_t)n * 32 + T - 1) / T);   // wave-per-node
  const int gE32 = (int)(((size_t)E * 32 + T - 1) / T);   // wave-per-edge
  const int gGemm = (n + 127) / 128;

  // normalization: deg = 1 (self loop) + in-degree; dis = deg^{-1/2}
  gcn_fill_f32<<<gN, T, 0, stream>>>(dis, 1.0f, n);
  gcn_deg_count<<<gE, T, 0, stream>>>(dis, dst, E);
  gcn_rsqrt_ip<<<gN, T, 0, stream>>>(dis, n);

  // layer 0: z0 = x @ W0 (outer product), aggregate, bias+relu
  gcn_layer0<<<gNH, T, 0, stream>>>(x, W0, wsA, n);
  gcn_agg_self<<<gN32, T, 0, stream>>>(wsA, wsB, dis, n);
  gcn_agg_edges<<<gE32, T, 0, stream>>>(wsA, wsB, src, dst, dis, E);
  gcn_bias_relu<<<gNH, T, 0, stream>>>(wsB, b0, n);

  // layer 1: WMMA GEMM, aggregate, bias+relu
  gcn_gemm_wmma<<<gGemm, T, 0, stream>>>(wsB, W1, wsA, n);
  gcn_agg_self<<<gN32, T, 0, stream>>>(wsA, wsB, dis, n);
  gcn_agg_edges<<<gE32, T, 0, stream>>>(wsA, wsB, src, dst, dis, E);
  gcn_bias_relu<<<gNH, T, 0, stream>>>(wsB, b1, n);

  // layer 2
  gcn_gemm_wmma<<<gGemm, T, 0, stream>>>(wsB, W2, wsA, n);
  gcn_agg_self<<<gN32, T, 0, stream>>>(wsA, wsB, dis, n);
  gcn_agg_edges<<<gE32, T, 0, stream>>>(wsA, wsB, src, dst, dis, E);
  gcn_bias_relu<<<gNH, T, 0, stream>>>(wsB, b2, n);

  // output head: 128 -> 1 dot, aggregate (H=1), sigmoid
  gcn_dot_out<<<gN32, T, 0, stream>>>(wsB, Wo, zo, n);
  gcn_acc_self<<<gN, T, 0, stream>>>(zo, acc, dis, n);
  gcn_acc_edges<<<gE, T, 0, stream>>>(zo, acc, src, dst, dis, E);
  gcn_sigmoid<<<gN, T, 0, stream>>>(acc, bo, out, n);
}
